// EdgeGenerator_62809601736824
// MI455X (gfx1250) — matrix operand order repaired and checked by reading, not compiled, not run on hardware
//
#include <hip/hip_runtime.h>
#include <hip/hip_bf16.h>
#include <math.h>

// ---------------------------------------------------------------------------
// EdgeGenerator GNN for MI455X (gfx1250, wave32, WMMA f16 -> f32)
//
// Pipeline:
//   0) transpose+pad all MLP weights to f16  wT[nout][Kpad]   (one-time)
//   1) node encoder  (3 -> 128 -> 128, tiny, VALU)
//   2) edge RBF features (E x 16, f16)
//   3) per layer l in 0..2:
//        zero agg
//        edge kernel: ASYNC global->LDS gather of h[src],h[dst],rbf,
//                     WMMA 288->128 (silu) -> LDS -> WMMA 128->128
//                     -> atomicAdd scatter to agg[dst]
//        node kernel: [h, agg] -> WMMA 256->128 (silu) -> WMMA 128->128
//                     -> residual + layernorm -> h (f32 + f16 copies)
//   4) pair head: gather h[u],h[v],r_pair -> WMMA 288->128 (silu) -> dot 128->1
// ---------------------------------------------------------------------------

typedef __attribute__((ext_vector_type(16))) _Float16 v16h;
typedef __attribute__((ext_vector_type(8)))  _Float16 v8h;
typedef __attribute__((ext_vector_type(8)))  float    v8f;

#define DH 128
#define NRBF 16

__device__ __forceinline__ float silu_f(float x) {
    return x * (1.0f / (1.0f + __expf(-x)));
}

__device__ __forceinline__ v8f zero8() {
    v8f z;
#pragma unroll
    for (int i = 0; i < 8; ++i) z[i] = 0.0f;
    return z;
}

// WMMA f16 A(16x32) * B(32x16) + C(16x16 f32)
__device__ __forceinline__ v8f wmma_step(v16h a, v16h b, v8f c) {
    return __builtin_amdgcn_wmma_f32_16x16x32_f16(
        /*neg_a=*/false, a, /*neg_b=*/false, b,
        /*c_mod=*/(short)0, c, /*reuse_a=*/false, /*reuse_b=*/false);
}

// Async 16-byte global -> LDS copy (gfx1250 GLOBAL_LOAD_ASYNC_TO_LDS_B128).
// LDS destination operand is the low 32 bits of the flat pointer
// (per ISA 10.2).
__device__ __forceinline__ void async_g2l_b128(void* lds_ptr, const void* gptr) {
    unsigned loff = (unsigned)(unsigned long long)lds_ptr;
    asm volatile("global_load_async_to_lds_b128 %0, %1, off"
                 :
                 : "v"(loff), "v"(gptr)
                 : "memory");
}

__device__ __forceinline__ void wait_async0() {
    asm volatile("s_wait_asynccnt 0x0" ::: "memory");
}

// A fragment from LDS, row-major [rows][strideK] halves.
// 16-bit A 16x32 layout: lane m=lane&15, g=lane>>4; element e -> K = e + 8*(e>=8) + 8*g
// => two contiguous 8-half chunks at kBase+8g and kBase+8g+16.
__device__ __forceinline__ v16h load_a_lds(const _Float16* base, int strideK,
                                           int rowBase, int kBase, int lane) {
    int m = lane & 15, g = lane >> 4;
    const _Float16* p = base + (rowBase + m) * strideK + kBase + 8 * g;
    v8h lo = *(const v8h*)p;
    v8h hi = *(const v8h*)(p + 16);
    return __builtin_shufflevector(lo, hi, 0, 1, 2, 3, 4, 5, 6, 7,
                                   8, 9, 10, 11, 12, 13, 14, 15);
}

// B fragment from global transposed weights wT[nout][Kpad] halves.
// 16-bit B 32x16 layout (same K order as A): element e -> K = e + 8*(e>=8) + 8*g
// => two contiguous 8-half chunks at kBase+8g and kBase+8g+16 of one output column.
__device__ __forceinline__ v16h load_b_glb(const _Float16* wT, int Kpad,
                                           int colTile, int kBase, int lane) {
    int n = (lane & 15) + 16 * colTile;
    int g = lane >> 4;
    const _Float16* p = wT + (size_t)n * Kpad + kBase + 8 * g;
    v8h lo = *(const v8h*)p;
    v8h hi = *(const v8h*)(p + 16);
    return __builtin_shufflevector(lo, hi, 0, 1, 2, 3, 4, 5, 6, 7,
                                   8, 9, 10, 11, 12, 13, 14, 15);
}

// ---------------------------------------------------------------------------
// Weight transpose+pad:  src fp32 (K, Nout) row-major -> dst f16 (Nout, Kpad)
// ---------------------------------------------------------------------------
__global__ void transpose_w_kernel(const float* __restrict__ src,
                                   _Float16* __restrict__ dst,
                                   int K, int Nout, int Kpad) {
    int i = blockIdx.x * 256 + threadIdx.x;
    if (i >= Nout * Kpad) return;
    int n = i / Kpad, k = i - n * Kpad;
    float v = (k < K) ? src[(size_t)k * Nout + n] : 0.0f;
    dst[i] = (_Float16)v;
}

// ---------------------------------------------------------------------------
// Node encoder: h = silu(x @ w1 + b1) @ w2 + b2    (3->128->128)
// One block of 128 threads per node.
// ---------------------------------------------------------------------------
__global__ __launch_bounds__(128) void node_encode_kernel(
    const float* __restrict__ xyr, int N,
    const float* __restrict__ w1, const float* __restrict__ b1,
    const float* __restrict__ w2, const float* __restrict__ b2,
    float* __restrict__ hF32, _Float16* __restrict__ hF16) {
    __shared__ float sH[DH];
    int nd = blockIdx.x;
    int tid = threadIdx.x;
    float x0 = xyr[nd * 3 + 0], x1 = xyr[nd * 3 + 1], x2 = xyr[nd * 3 + 2];
    float h1 = silu_f(x0 * w1[tid] + x1 * w1[DH + tid] + x2 * w1[2 * DH + tid] + b1[tid]);
    sH[tid] = h1;
    __syncthreads();
    float h2 = b2[tid];
#pragma unroll 4
    for (int k = 0; k < DH; ++k) h2 += sH[k] * w2[k * DH + tid];
    hF32[(size_t)nd * DH + tid] = h2;
    hF16[(size_t)nd * DH + tid] = (_Float16)h2;
}

// ---------------------------------------------------------------------------
// Edge RBF features (E x 16 f16)
// ---------------------------------------------------------------------------
__global__ void rbf_kernel(const float* __restrict__ xyr,
                           const int* __restrict__ eidx, int E,
                           _Float16* __restrict__ rbf) {
    int e = blockIdx.x * 256 + threadIdx.x;
    if (e >= E) return;
    int s = eidx[e], d = eidx[E + e];
    float dx = xyr[s * 3 + 0] - xyr[d * 3 + 0];
    float dy = xyr[s * 3 + 1] - xyr[d * 3 + 1];
    float r = sqrtf(dx * dx + dy * dy + 1e-8f);
    const float RMAX = 1.41421356237309515f;
    const float width = RMAX / (NRBF - 1);
    const float gamma = 0.5f / (width * width);
#pragma unroll
    for (int i = 0; i < NRBF; ++i) {
        float t = r - width * (float)i;
        rbf[(size_t)e * NRBF + i] = (_Float16)__expf(-gamma * t * t);
    }
}

__global__ void zero_f32_kernel(float* __restrict__ p, size_t n) {
    size_t i = (size_t)blockIdx.x * 256 + threadIdx.x;
    if (i < n) p[i] = 0.0f;
}

// ---------------------------------------------------------------------------
// Edge message kernel (WMMA + async gather).
// Block: 256 threads (8 wave32), 64 edges. Wave w owns output cols [16w,16w+16).
// K layout of m_in: [ h[src] 0..127 | h[dst] 128..255 | rbf 256..271 | pad ..287 ]
// Staging is done with GLOBAL_LOAD_ASYNC_TO_LDS_B128 (ASYNCcnt), all transfers
// are aligned 16B chunks: h rows = 16 x b128, rbf rows = 2 x b128.
// ---------------------------------------------------------------------------
__global__ __launch_bounds__(256) void edge_msg_kernel(
    const _Float16* __restrict__ hF16, const _Float16* __restrict__ rbf,
    const int* __restrict__ eidx, int E,
    const _Float16* __restrict__ w1T, const float* __restrict__ b1,
    const _Float16* __restrict__ w2T, const float* __restrict__ b2,
    float* __restrict__ agg) {
    __shared__ __align__(32) _Float16 sIn[64 * 288];
    __shared__ __align__(32) _Float16 sHid[64 * DH];
    __shared__ int sSrc[64];
    __shared__ int sDst[64];

    const int tid = threadIdx.x;
    const int base = blockIdx.x * 64;

    if (tid < 64) {
        int e = base + tid;
        int s = 0, d = 0;
        if (e < E) { s = eidx[e]; d = eidx[E + e]; }
        sSrc[tid] = s;
        sDst[tid] = d;
    }
    __syncthreads();

    // --- Async stage h[src] | h[dst]: 64 rows x 32 b128 chunks (16 src + 16 dst).
    for (int i = tid; i < 64 * 32; i += 256) {
        int row = i >> 5, c = i & 31;
        if (base + row < E) {
            const _Float16* g;
            _Float16* l;
            if (c < 16) {
                g = hF16 + (size_t)sSrc[row] * DH + c * 8;
                l = &sIn[row * 288 + c * 8];
            } else {
                g = hF16 + (size_t)sDst[row] * DH + (c - 16) * 8;
                l = &sIn[row * 288 + DH + (c - 16) * 8];
            }
            async_g2l_b128(l, g);
        }
    }
    // --- Async stage rbf: 64 rows x 2 b128 chunks.
    for (int i = tid; i < 64 * 2; i += 256) {
        int row = i >> 1, c = i & 1;
        int e = base + row;
        if (e < E) {
            async_g2l_b128(&sIn[row * 288 + 2 * DH + c * 8],
                           rbf + (size_t)e * NRBF + c * 8);
        }
    }
    // --- Zero the K-pad columns [272, 288) with plain LDS stores.
    for (int i = tid; i < 64 * 16; i += 256) {
        int row = i >> 4, k = i & 15;
        sIn[row * 288 + 272 + k] = (_Float16)0.0f;
    }
    wait_async0();
    __syncthreads();

    const int w = tid >> 5, lane = tid & 31;
    const int n = lane & 15, g = lane >> 4;
    const int col = 16 * w + n;

    // Layer 1: (64x288) @ (288x128) -> silu -> sHid
    v8f acc[4];
#pragma unroll
    for (int t = 0; t < 4; ++t) acc[t] = zero8();
#pragma unroll
    for (int c = 0; c < 9; ++c) {
        v16h b = load_b_glb(w1T, 288, w, 32 * c, lane);
#pragma unroll
        for (int t = 0; t < 4; ++t) {
            v16h a = load_a_lds(sIn, 288, 16 * t, 32 * c, lane);
            acc[t] = wmma_step(a, b, acc[t]);
        }
    }
    {
        float bias = b1[col];
#pragma unroll
        for (int t = 0; t < 4; ++t)
#pragma unroll
            for (int r = 0; r < 8; ++r) {
                int row = 16 * t + r + 8 * g;
                sHid[row * DH + col] = (_Float16)silu_f(acc[t][r] + bias);
            }
    }
    __syncthreads();

    // Layer 2: (64x128) @ (128x128)
    v8f acc2[4];
#pragma unroll
    for (int t = 0; t < 4; ++t) acc2[t] = zero8();
#pragma unroll
    for (int c = 0; c < 4; ++c) {
        v16h b = load_b_glb(w2T, DH, w, 32 * c, lane);
#pragma unroll
        for (int t = 0; t < 4; ++t) {
            v16h a = load_a_lds(sHid, DH, 16 * t, 32 * c, lane);
            acc2[t] = wmma_step(a, b, acc2[t]);
        }
    }

    // Fused segment_sum: scatter-add message into agg[dst].
    {
        float bias2 = b2[col];
#pragma unroll
        for (int t = 0; t < 4; ++t)
#pragma unroll
            for (int r = 0; r < 8; ++r) {
                int row = 16 * t + r + 8 * g;
                int e = base + row;
                if (e < E) {
                    atomicAdd(&agg[(size_t)sDst[row] * DH + col], acc2[t][r] + bias2);
                }
            }
    }
}

// ---------------------------------------------------------------------------
// Node update: h_up = mlp2([h, agg]); h = layernorm(h + h_up)
// Block: 256 threads (8 waves), 64 nodes.
// ---------------------------------------------------------------------------
__global__ __launch_bounds__(256) void node_update_kernel(
    float* __restrict__ hF32, _Float16* __restrict__ hF16,
    const float* __restrict__ agg, int N,
    const _Float16* __restrict__ w1T, const float* __restrict__ b1,
    const _Float16* __restrict__ w2T, const float* __restrict__ b2,
    const float* __restrict__ lng, const float* __restrict__ lnb) {
    __shared__ __align__(32) char smem[64 * 256 * 2 + 64 * DH * 2];
    _Float16* sIn  = (_Float16*)smem;                  // 64 x 256 f16
    _Float16* sHid = (_Float16*)(smem + 64 * 256 * 2); // 64 x 128 f16
    float*    sUp  = (float*)smem;                     // overlaps sIn: 64 x 128 f32

    const int tid = threadIdx.x;
    const int base = blockIdx.x * 64;

    for (int i = tid; i < 64 * 256; i += 256) {
        int row = i >> 8, k = i & 255;
        int nd = base + row;
        _Float16 v = (_Float16)0.0f;
        if (nd < N) {
            v = (k < DH) ? hF16[(size_t)nd * DH + k]
                         : (_Float16)agg[(size_t)nd * DH + (k - DH)];
        }
        sIn[i] = v;
    }
    __syncthreads();

    const int w = tid >> 5, lane = tid & 31;
    const int n = lane & 15, g = lane >> 4;
    const int col = 16 * w + n;

    v8f acc[4];
#pragma unroll
    for (int t = 0; t < 4; ++t) acc[t] = zero8();
#pragma unroll
    for (int c = 0; c < 8; ++c) {
        v16h b = load_b_glb(w1T, 256, w, 32 * c, lane);
#pragma unroll
        for (int t = 0; t < 4; ++t) {
            v16h a = load_a_lds(sIn, 256, 16 * t, 32 * c, lane);
            acc[t] = wmma_step(a, b, acc[t]);
        }
    }
    {
        float bias = b1[col];
#pragma unroll
        for (int t = 0; t < 4; ++t)
#pragma unroll
            for (int r = 0; r < 8; ++r) {
                int row = 16 * t + r + 8 * g;
                sHid[row * DH + col] = (_Float16)silu_f(acc[t][r] + bias);
            }
    }
    __syncthreads();  // all waves done reading sIn & writing sHid

    v8f acc2[4];
#pragma unroll
    for (int t = 0; t < 4; ++t) acc2[t] = zero8();
#pragma unroll
    for (int c = 0; c < 4; ++c) {
        v16h b = load_b_glb(w2T, DH, w, 32 * c, lane);
#pragma unroll
        for (int t = 0; t < 4; ++t) {
            v16h a = load_a_lds(sHid, DH, 16 * t, 32 * c, lane);
            acc2[t] = wmma_step(a, b, acc2[t]);
        }
    }
    // h_up (f32) into sUp (reuses the sIn region; safe after barrier above).
    {
        float bias2 = b2[col];
#pragma unroll
        for (int t = 0; t < 4; ++t)
#pragma unroll
            for (int r = 0; r < 8; ++r) {
                int row = 16 * t + r + 8 * g;
                sUp[row * DH + col] = acc2[t][r] + bias2;
            }
    }
    __syncthreads();

    // Residual + layernorm; one thread per node row.
    if (tid < 64) {
        int nd = base + tid;
        if (nd < N) {
            float sum = 0.0f, sq = 0.0f;
            for (int k = 0; k < DH; ++k) {
                float v = hF32[(size_t)nd * DH + k] + sUp[tid * DH + k];
                sum += v;
                sq += v * v;
            }
            float mu = sum * (1.0f / DH);
            float var = sq * (1.0f / DH) - mu * mu;
            float inv = rsqrtf(var + 1e-5f);
            for (int k = 0; k < DH; ++k) {
                float v = hF32[(size_t)nd * DH + k] + sUp[tid * DH + k];
                float o = (v - mu) * inv * lng[k] + lnb[k];
                hF32[(size_t)nd * DH + k] = o;
                hF16[(size_t)nd * DH + k] = (_Float16)o;
            }
        }
    }
}

// ---------------------------------------------------------------------------
// Pair head: feat = [h[u], h[v], r_pair(4)] (K=260 pad 288) -> 128 silu -> 1
// ---------------------------------------------------------------------------
__global__ __launch_bounds__(256) void pair_head_kernel(
    const _Float16* __restrict__ hF16, const float* __restrict__ xyr,
    const int* __restrict__ pairs, int P,
    const _Float16* __restrict__ w1T, const float* __restrict__ b1,
    const float* __restrict__ w2, const float* __restrict__ b2,
    float* __restrict__ out) {
    __shared__ __align__(32) _Float16 sIn[64 * 288];
    __shared__ __align__(32) _Float16 sHid[64 * DH];
    __shared__ int sU[64];
    __shared__ int sV[64];

    const int tid = threadIdx.x;
    const int base = blockIdx.x * 64;

    if (tid < 64) {
        int p = base + tid;
        int u = 0, v = 0;
        if (p < P) { u = pairs[2 * p]; v = pairs[2 * p + 1]; }
        sU[tid] = u;
        sV[tid] = v;
    }
    __syncthreads();

    for (int i = tid; i < 64 * 288; i += 256) {
        int row = i / 288, k = i - row * 288;
        int p = base + row;
        _Float16 val = (_Float16)0.0f;
        if (p < P) {
            int u = sU[row], v = sV[row];
            if (k < DH) {
                val = hF16[(size_t)u * DH + k];
            } else if (k < 2 * DH) {
                val = hF16[(size_t)v * DH + (k - DH)];
            } else if (k < 2 * DH + 4) {
                float ru = xyr[u * 3 + 2], rv = xyr[v * 3 + 2];
                if (k == 2 * DH + 0) val = (_Float16)ru;
                else if (k == 2 * DH + 1) val = (_Float16)rv;
                else if (k == 2 * DH + 2) {
                    float dx = xyr[u * 3 + 0] - xyr[v * 3 + 0];
                    float dy = xyr[u * 3 + 1] - xyr[v * 3 + 1];
                    val = (_Float16)sqrtf(dx * dx + dy * dy + 1e-8f);
                } else {
                    val = (_Float16)fabsf(ru - rv);
                }
            }
        }
        sIn[i] = val;
    }
    __syncthreads();

    const int w = tid >> 5, lane = tid & 31;
    const int n = lane & 15, g = lane >> 4;
    const int col = 16 * w + n;

    v8f acc[4];
#pragma unroll
    for (int t = 0; t < 4; ++t) acc[t] = zero8();
#pragma unroll
    for (int c = 0; c < 9; ++c) {
        v16h b = load_b_glb(w1T, 288, w, 32 * c, lane);
#pragma unroll
        for (int t = 0; t < 4; ++t) {
            v16h a = load_a_lds(sIn, 288, 16 * t, 32 * c, lane);
            acc[t] = wmma_step(a, b, acc[t]);
        }
    }
    {
        float bias = b1[col];
#pragma unroll
        for (int t = 0; t < 4; ++t)
#pragma unroll
            for (int r = 0; r < 8; ++r) {
                int row = 16 * t + r + 8 * g;
                sHid[row * DH + col] = (_Float16)silu_f(acc[t][r] + bias);
            }
    }
    __syncthreads();

    // Final 128 -> 1 dot per pair row.
    if (tid < 64) {
        int p = base + tid;
        if (p < P) {
            float s = b2[0];
#pragma unroll 4
            for (int k = 0; k < DH; ++k) s += (float)sHid[tid * DH + k] * w2[k];
            out[p] = s;
        }
    }
}

// ---------------------------------------------------------------------------
// Host-side orchestration
// ---------------------------------------------------------------------------
extern "C" void kernel_launch(void* const* d_in, const int* in_sizes, int n_in,
                              void* d_out, int out_size, void* d_ws, size_t ws_size,
                              hipStream_t stream) {
    const float* xyr     = (const float*)d_in[0];
    const int*   eidx    = (const int*)d_in[1];
    const int*   pairs   = (const int*)d_in[2];
    const float* node_w1 = (const float*)d_in[3];
    const float* node_b1 = (const float*)d_in[4];
    const float* node_w2 = (const float*)d_in[5];
    const float* node_b2 = (const float*)d_in[6];
    const float* pm_w1   = (const float*)d_in[7];
    const float* pm_b1   = (const float*)d_in[8];
    const float* pm_w2   = (const float*)d_in[9];
    const float* pm_b2   = (const float*)d_in[10];
    const float* ph_w1   = (const float*)d_in[11];
    const float* ph_b1   = (const float*)d_in[12];
    const float* ph_w2   = (const float*)d_in[13];
    const float* ph_b2   = (const float*)d_in[14];
    const float* ln_g    = (const float*)d_in[15];
    const float* ln_b    = (const float*)d_in[16];
    const float* eh_w1   = (const float*)d_in[17];
    const float* eh_b1   = (const float*)d_in[18];
    const float* eh_w2   = (const float*)d_in[19];
    const float* eh_b2   = (const float*)d_in[20];

    const int N = in_sizes[0] / 3;
    const int E = in_sizes[1] / 2;
    const int P = in_sizes[2] / 2;
    const int L = 3;

    // Workspace carve-out (256B aligned).
    char* ws = (char*)d_ws;
    size_t off = 0;
    auto carve = [&](size_t bytes) -> char* {
        char* p = ws + off;
        off = (off + bytes + 255) & ~(size_t)255;
        return p;
    };
    float*    hF32   = (float*)carve((size_t)N * DH * 4);
    _Float16* hF16   = (_Float16*)carve((size_t)N * DH * 2);
    float*    agg    = (float*)carve((size_t)N * DH * 4);
    _Float16* rbf    = (_Float16*)carve((size_t)E * NRBF * 2);
    _Float16* pm_w1T = (_Float16*)carve((size_t)L * DH * 288 * 2);
    _Float16* pm_w2T = (_Float16*)carve((size_t)L * DH * DH * 2);
    _Float16* ph_w1T = (_Float16*)carve((size_t)L * DH * 256 * 2);
    _Float16* ph_w2T = (_Float16*)carve((size_t)L * DH * DH * 2);
    _Float16* eh_w1T = (_Float16*)carve((size_t)DH * 288 * 2);
    (void)ws_size; (void)n_in; (void)out_size;

    auto grid1 = [](size_t n) { return dim3((unsigned)((n + 255) / 256)); };

    // 0) one-time weight transpose+pad to f16.
    for (int l = 0; l < L; ++l) {
        transpose_w_kernel<<<grid1((size_t)DH * 288), 256, 0, stream>>>(
            pm_w1 + (size_t)l * 272 * DH, pm_w1T + (size_t)l * DH * 288, 272, DH, 288);
        transpose_w_kernel<<<grid1((size_t)DH * DH), 256, 0, stream>>>(
            pm_w2 + (size_t)l * DH * DH, pm_w2T + (size_t)l * DH * DH, DH, DH, DH);
        transpose_w_kernel<<<grid1((size_t)DH * 256), 256, 0, stream>>>(
            ph_w1 + (size_t)l * 256 * DH, ph_w1T + (size_t)l * DH * 256, 256, DH, 256);
        transpose_w_kernel<<<grid1((size_t)DH * DH), 256, 0, stream>>>(
            ph_w2 + (size_t)l * DH * DH, ph_w2T + (size_t)l * DH * DH, DH, DH, DH);
    }
    transpose_w_kernel<<<grid1((size_t)DH * 288), 256, 0, stream>>>(
        eh_w1, eh_w1T, 260, DH, 288);

    // 1) node encoder, 2) rbf.
    node_encode_kernel<<<dim3(N), 128, 0, stream>>>(
        xyr, N, node_w1, node_b1, node_w2, node_b2, hF32, hF16);
    rbf_kernel<<<grid1((size_t)E), 256, 0, stream>>>(xyr, eidx, E, rbf);

    // 3) message-passing layers.
    for (int l = 0; l < L; ++l) {
        zero_f32_kernel<<<grid1((size_t)N * DH), 256, 0, stream>>>(agg, (size_t)N * DH);
        edge_msg_kernel<<<dim3((E + 63) / 64), 256, 0, stream>>>(
            hF16, rbf, eidx, E,
            pm_w1T + (size_t)l * DH * 288, pm_b1 + (size_t)l * DH,
            pm_w2T + (size_t)l * DH * DH,  pm_b2 + (size_t)l * DH, agg);
        node_update_kernel<<<dim3((N + 63) / 64), 256, 0, stream>>>(
            hF32, hF16, agg, N,
            ph_w1T + (size_t)l * DH * 256, ph_b1 + (size_t)l * DH,
            ph_w2T + (size_t)l * DH * DH,  ph_b2 + (size_t)l * DH,
            ln_g + (size_t)l * DH, ln_b + (size_t)l * DH);
    }

    // 4) pair head.
    pair_head_kernel<<<dim3((P + 63) / 64), 256, 0, stream>>>(
        hF16, xyr, pairs, P, eh_w1T, eh_b1, eh_w2, eh_b2, (float*)d_out);
}
